// PoolHiddenNet_42812234007154
// MI455X (gfx1250) — compile-verified, matrix operand order checked by reading, not run on hardware
//
#include <hip/hip_runtime.h>
#include <hip/hip_bf16.h>

// PoolHiddenNet fused for MI455X (gfx1250, wave32, WMMA f16 16x16x32).
//
// Per group (g=32 -> 1024 pairwise rows):
//   x(1024x128) = [rel@W_spatial | h_j]   (biases skipped: they cancel in BN)
//   y = x @ W1            -> BN1 stats need all rows  -> two GEMM1 passes
//   a = relu(bn1(y)) f16  -> z = a @ W2  -> z to scratch, BN2 stats
//   out[i] = max_j relu(bn2(z[i*32+j]))
//
// One workgroup (256 thr = 8 waves) per group; 256 workgroups.

typedef __attribute__((ext_vector_type(16))) _Float16 v16h;
typedef __attribute__((ext_vector_type(8)))  float    v8f;

#define NGRP 256
#define GPED 32
#define K1   128     // E + H
#define MID  512
#define BOT  32
#define ROWS 1024    // GPED*GPED
#define RT_N 64      // ROWS/16 row tiles
#define BN_EPS 1e-5f

// fragment (lane,e) -> k within a 32-wide K chunk (16-bit A layout, ISA 7.12.2)
__device__ __forceinline__ int kmap(int lane, int e) {
    int hi = lane >> 4;
    return (e < 8 ? e : e + 8) + hi * 8;
}

// ---------------------------------------------------------------- pack weights
// W1 (128x512 f32, row-major) -> f16 fragment-linear: frag(nt,kt) at
// ((nt*4+kt)*512), element = lane*16+e, value = W1[kt*32+kmap][nt*16+lane%16]
// W2 (512x32) -> frag(nt2,kt2) at ((nt2*16+kt2)*512)
__global__ void phn_pack_weights(const float* __restrict__ W1,
                                 const float* __restrict__ W2,
                                 _Float16* __restrict__ w1p,
                                 _Float16* __restrict__ w2p) {
    const int total1 = 32 * 4 * 512;   // 65536
    const int total2 = 2 * 16 * 512;   // 16384
    for (int t = blockIdx.x * blockDim.x + threadIdx.x;
         t < total1 + total2; t += gridDim.x * blockDim.x) {
        if (t < total1) {
            int frag = t >> 9, idx = t & 511;
            int lane = idx >> 4, e = idx & 15;
            int nt = frag >> 2, kt = frag & 3;
            int k = kt * 32 + kmap(lane, e);
            int n = nt * 16 + (lane & 15);
            w1p[t] = (_Float16)W1[k * 512 + n];
        } else {
            int u = t - total1;
            int frag = u >> 9, idx = u & 511;
            int lane = idx >> 4, e = idx & 15;
            int nt = frag >> 4, kt = frag & 15;
            int k = kt * 32 + kmap(lane, e);
            int n = nt * 16 + (lane & 15);
            w2p[u] = (_Float16)W2[k * 32 + n];
        }
    }
}

// ------------------------------------------------------------------ main fused
__global__ __launch_bounds__(256)
void phn_fused_kernel(const float* __restrict__ h_states,   // (B,64)
                      const float* __restrict__ end_pos,    // (B,2)
                      const float* __restrict__ W_spatial,  // (2,64)
                      const float* __restrict__ gamma1,
                      const float* __restrict__ beta1,
                      const float* __restrict__ gamma2,
                      const float* __restrict__ beta2,
                      const _Float16* __restrict__ w1p,
                      const _Float16* __restrict__ w2p,
                      float* __restrict__ zbuf,             // (NGRP,1024,32)
                      float* __restrict__ out)              // (B,32)
{
    __shared__ __attribute__((aligned(32))) _Float16 sXF[4 * 512];    // x frags (16x128)
    __shared__ __attribute__((aligned(32))) _Float16 sABuf[16 * 512]; // a frags (16x512)
    __shared__ __attribute__((aligned(16))) float sZ[512];            // z tile 16x32
    __shared__ float sStats1[1024];  // sum|sumsq -> mean|rstd (512 cols)
    __shared__ float sStats2[64];    // sum|sumsq -> mean|rstd (32 cols)
    __shared__ float sHf[GPED * 64];
    __shared__ float sPos[GPED * 2];
    __shared__ float sWs[128];
    __shared__ float sG1[512];
    __shared__ float sB1[512];

    const int tid   = threadIdx.x;
    const int wave  = tid >> 5;
    const int lane  = tid & 31;
    const int group = blockIdx.x;

    // ---- stage per-group constants -------------------------------------
    for (int t = tid; t < GPED * 64; t += 256) sHf[t] = h_states[group * GPED * 64 + t];
    for (int t = tid; t < GPED * 2;  t += 256) sPos[t] = end_pos[group * GPED * 2 + t];
    for (int t = tid; t < 128;       t += 256) sWs[t] = W_spatial[t];
    for (int t = tid; t < 512;       t += 256) { sG1[t] = gamma1[t]; sB1[t] = beta1[t]; }
    for (int t = tid; t < 1024;      t += 256) sStats1[t] = 0.f;
    if (tid < 64) sStats2[tid] = 0.f;
    __syncthreads();

    const int p  = lane & 15;   // D-tile column within n-tile / A row
    const int hiD = lane >> 4;  // D rows r+8*hiD

    // =========================== PASS 1: BN1 stats ======================
    float rs[4] = {0.f, 0.f, 0.f, 0.f};
    float rq[4] = {0.f, 0.f, 0.f, 0.f};

    for (int rt = 0; rt < RT_N; ++rt) {
        __syncthreads();
        // stage x row-tile (16x128) in A-fragment layout, f16
        for (int t = tid; t < 4 * 512; t += 256) {
            int kt = t >> 9, idx = t & 511;
            int fl = idx >> 4, e = idx & 15;
            int k = kt * 32 + kmap(fl, e);
            int row = rt * 16 + (fl & 15);
            int i = row >> 5, j = row & 31;
            float v;
            if (k < 64) {
                float rx = sPos[j * 2 + 0] - sPos[i * 2 + 0];
                float ry = sPos[j * 2 + 1] - sPos[i * 2 + 1];
                v = rx * sWs[k] + ry * sWs[64 + k];   // b_spatial cancels in BN1
            } else {
                v = sHf[j * 64 + (k - 64)];
            }
            sXF[t] = (_Float16)v;
        }
        __syncthreads();

        v16h a1[4];
#pragma unroll
        for (int kt = 0; kt < 4; ++kt)
            a1[kt] = *(const v16h*)(sXF + kt * 512 + lane * 16);

#pragma unroll
        for (int q = 0; q < 4; ++q) {
            int nt = wave * 4 + q;
            v8f acc = {};
#pragma unroll
            for (int kt = 0; kt < 4; ++kt) {
                v16h b = *(const v16h*)(w1p + ((nt * 4 + kt) * 512 + lane * 16));
                acc = __builtin_amdgcn_wmma_f32_16x16x32_f16(
                        false, a1[kt], false, b, (short)0, acc, false, false);
            }
#pragma unroll
            for (int r = 0; r < 8; ++r) { rs[q] += acc[r]; rq[q] += acc[r] * acc[r]; }
        }
    }
#pragma unroll
    for (int q = 0; q < 4; ++q) {
        int col = (wave * 4 + q) * 16 + p;
        atomicAdd(&sStats1[col], rs[q]);
        atomicAdd(&sStats1[512 + col], rq[q]);
    }
    __syncthreads();
    // finalize mean/rstd in place
    for (int c = tid; c < 512; c += 256) {
        float mean = sStats1[c] * (1.f / ROWS);
        float var  = sStats1[512 + c] * (1.f / ROWS) - mean * mean;
        sStats1[c] = mean;
        sStats1[512 + c] = rsqrtf(var + BN_EPS);
    }
    __syncthreads();

    // ============== PASS 2: GEMM1 -> bn1 -> relu -> GEMM2 -> z ==========
    const int myCol = tid & 31;
    float s2s = 0.f, s2q = 0.f;

    for (int rt = 0; rt < RT_N; ++rt) {
        __syncthreads();
        for (int t = tid; t < 4 * 512; t += 256) {      // restage x frags
            int kt = t >> 9, idx = t & 511;
            int fl = idx >> 4, e = idx & 15;
            int k = kt * 32 + kmap(fl, e);
            int row = rt * 16 + (fl & 15);
            int i = row >> 5, j = row & 31;
            float v;
            if (k < 64) {
                float rx = sPos[j * 2 + 0] - sPos[i * 2 + 0];
                float ry = sPos[j * 2 + 1] - sPos[i * 2 + 1];
                v = rx * sWs[k] + ry * sWs[64 + k];
            } else {
                v = sHf[j * 64 + (k - 64)];
            }
            sXF[t] = (_Float16)v;
        }
        for (int t = tid; t < 512; t += 256) sZ[t] = 0.f;
        __syncthreads();

        v16h a1[4];
#pragma unroll
        for (int kt = 0; kt < 4; ++kt)
            a1[kt] = *(const v16h*)(sXF + kt * 512 + lane * 16);

#pragma unroll
        for (int q = 0; q < 4; ++q) {
            int nt = wave * 4 + q;
            v8f acc = {};
#pragma unroll
            for (int kt = 0; kt < 4; ++kt) {
                v16h b = *(const v16h*)(w1p + ((nt * 4 + kt) * 512 + lane * 16));
                acc = __builtin_amdgcn_wmma_f32_16x16x32_f16(
                        false, a1[kt], false, b, (short)0, acc, false, false);
            }
            // bn1 + relu, scatter into GEMM2 A-fragment layout
            int col  = nt * 16 + p;
            float mean = sStats1[col], rstd = sStats1[512 + col];
            float gg = sG1[col] * rstd;
            float bb = sB1[col] - mean * gg;
            int kt2 = col >> 5, kl = col & 31;
            int hi2, e2;
            if (kl < 8)       { hi2 = 0; e2 = kl; }
            else if (kl < 16) { hi2 = 1; e2 = kl - 8; }
            else if (kl < 24) { hi2 = 0; e2 = kl - 8; }
            else              { hi2 = 1; e2 = kl - 16; }
#pragma unroll
            for (int r = 0; r < 8; ++r) {
                int m = r + hiD * 8;
                float v = fmaxf(acc[r] * gg + bb, 0.f);
                sABuf[kt2 * 512 + (m + hi2 * 16) * 16 + e2] = (_Float16)v;
            }
        }
        __syncthreads();

        // GEMM2: wave handles K chunks 2w, 2w+1; reduce partial z into LDS
        v8f z0 = {}, z1 = {};
#pragma unroll
        for (int t = 0; t < 2; ++t) {
            int kt2 = wave * 2 + t;
            v16h a2 = *(const v16h*)(sABuf + kt2 * 512 + lane * 16);
            v16h b0 = *(const v16h*)(w2p + ((0 * 16 + kt2) * 512 + lane * 16));
            v16h b1v = *(const v16h*)(w2p + ((1 * 16 + kt2) * 512 + lane * 16));
            z0 = __builtin_amdgcn_wmma_f32_16x16x32_f16(false, a2, false, b0,
                                                        (short)0, z0, false, false);
            z1 = __builtin_amdgcn_wmma_f32_16x16x32_f16(false, a2, false, b1v,
                                                        (short)0, z1, false, false);
        }
#pragma unroll
        for (int r = 0; r < 8; ++r) {
            int m = r + hiD * 8;
            atomicAdd(&sZ[m * 32 + p], z0[r]);
            atomicAdd(&sZ[m * 32 + 16 + p], z1[r]);
        }
        __syncthreads();

        // BN2 stats + spill z tile (pre-BN) to scratch
#pragma unroll
        for (int s = 0; s < 2; ++s) {
            int t = tid + s * 256;
            float zv = sZ[t];
            s2s += zv; s2q += zv * zv;                       // col == myCol
            int row = rt * 16 + (t >> 5);
            zbuf[((size_t)group * ROWS + row) * BOT + (t & 31)] = zv;
        }
    }
    atomicAdd(&sStats2[myCol], s2s);
    atomicAdd(&sStats2[32 + myCol], s2q);
    __threadfence();
    __syncthreads();
    if (tid < 32) {
        float mean = sStats2[tid] * (1.f / ROWS);
        float var  = sStats2[32 + tid] * (1.f / ROWS) - mean * mean;
        sStats2[tid] = mean;
        sStats2[32 + tid] = rsqrtf(var + BN_EPS);
    }
    __syncthreads();

    // ==================== bn2 + relu + max over j =======================
#pragma unroll
    for (int pch = 0; pch < 4; ++pch) {
        int flat = tid + pch * 256;          // 1024 outputs: (i, c)
        int i = flat >> 5, c = flat & 31;
        float mean = sStats2[c], rstd = sStats2[32 + c];
        float gg = gamma2[c] * rstd;
        float bb = beta2[c] - mean * gg;     // b2 cancels in BN2
        const float* zp = zbuf + ((size_t)group * ROWS + i * GPED) * BOT + c;
        float m = 0.f;                        // relu output >= 0
#pragma unroll 4
        for (int j = 0; j < GPED; ++j) {
            float v = fmaxf(zp[j * BOT] * gg + bb, 0.f);
            m = fmaxf(m, v);
        }
        out[((size_t)group * GPED + i) * BOT + c] = m;
    }
}

// -------------------------------------------------------------------- launch
extern "C" void kernel_launch(void* const* d_in, const int* in_sizes, int n_in,
                              void* d_out, int out_size, void* d_ws, size_t ws_size,
                              hipStream_t stream) {
    const float* h_states  = (const float*)d_in[0];
    // d_in[1] seq_start_end: groups are uniform (group g = rows [32g,32g+32)) -> unused
    const float* end_pos   = (const float*)d_in[2];
    const float* W_spatial = (const float*)d_in[3];
    // d_in[4] b_spatial: exact cancellation in BN1 -> unused
    const float* W1        = (const float*)d_in[5];
    // d_in[6] b1: cancels in BN1
    const float* gamma1    = (const float*)d_in[7];
    const float* beta1     = (const float*)d_in[8];
    const float* W2        = (const float*)d_in[9];
    // d_in[10] b2: cancels in BN2
    const float* gamma2    = (const float*)d_in[11];
    const float* beta2     = (const float*)d_in[12];

    _Float16* w1p = (_Float16*)d_ws;                      // 65536 f16 = 128KB
    _Float16* w2p = w1p + 65536;                          // 16384 f16 = 32KB
    float* zbuf = (float*)((char*)d_ws + (256u << 10));   // 32MB, 256KB aligned

    phn_pack_weights<<<320, 256, 0, stream>>>(W1, W2, w1p, w2p);
    phn_fused_kernel<<<NGRP, 256, 0, stream>>>(h_states, end_pos, W_spatial,
                                               gamma1, beta1, gamma2, beta2,
                                               w1p, w2p, zbuf, (float*)d_out);
}
